// SpookyNetAttention_26121991094378
// MI455X (gfx1250) — matrix operand order repaired and checked by reading, not compiled, not run on hardware
//
#include <hip/hip_runtime.h>

typedef __attribute__((ext_vector_type(16))) _Float16 v16h;
typedef __attribute__((ext_vector_type(8)))  float    v8f;
typedef __attribute__((ext_vector_type(4)))  float    v4f;

#define DQK      128
#define MFEAT    256
#define NT_TILES 16   // 256 / 16
#define KT_TILES 4    // 128 / 32
#define WPB      8    // waves per block (256 threads, wave32)
#define TPB      256
#define SPW      4    // strips per wave (grid-stride)

#define XSCALE      0.29730177875068026f   // 128^-0.25
#define HSCALE      0.04419417382415922f   // 1/(2*sqrt(128))
#define INV_SQRT_M  0.0625f                // 1/sqrt(256)
#define NEG_BIG    -3.0e38f

// ws float layout:
//  [0]                      maxK
//  [64 .. 64+255]           sumK (final)
//  [512 .. 512+N)           w[N]
//  [512+N .. 512+2N)        rowmaxQ[N]
//  [512+2N .. )             per-block sumK partials (nBlocks * 256)
#define WS_MAXK 0
#define WS_SUMK 64
#define WS_W    512

__device__ __forceinline__ void atomic_max_f32(float* addr, float val) {
  unsigned int* ua = (unsigned int*)addr;
  unsigned int cur = *ua;
  while (__uint_as_float(cur) < val) {
    unsigned int assumed = cur;
    cur = atomicCAS(ua, assumed, __float_as_uint(val));
    if (cur == assumed) break;
  }
}

__device__ __forceinline__ float redmax16(float v) {
  v = fmaxf(v, __shfl_xor(v, 1, 32));
  v = fmaxf(v, __shfl_xor(v, 2, 32));
  v = fmaxf(v, __shfl_xor(v, 4, 32));
  v = fmaxf(v, __shfl_xor(v, 8, 32));
  return v;
}
__device__ __forceinline__ float redsum16(float v) {
  v += __shfl_xor(v, 1, 32);
  v += __shfl_xor(v, 2, 32);
  v += __shfl_xor(v, 4, 32);
  v += __shfl_xor(v, 8, 32);
  return v;
}

// Load rows [row0, row0+16) of X as four 16x32 f16 A-tiles (CDNA5 16-bit A layout:
// lane<16 holds K {0..7,16..23}, lane>=16 holds K {8..15,24..31} of each 32-K tile,
// row m = lane&15). Also returns h[m] = sum(x^2) * HSCALE (valid in every lane).
__device__ __forceinline__ void load_A(const float* __restrict__ X, int row0, int lane,
                                       v16h (&a)[KT_TILES], float& hOut) {
  const int m  = lane & 15;
  const int co = (lane >> 4) << 1;  // chunk offset in float4 units (0 or 2)
  const v4f* r4 = (const v4f*)(X + (size_t)(row0 + m) * DQK);
  float ss = 0.0f;
#pragma unroll
  for (int kt = 0; kt < KT_TILES; ++kt) {
    v4f x0 = r4[kt * 8 + co + 0];
    v4f x1 = r4[kt * 8 + co + 1];
    v4f x2 = r4[kt * 8 + 4 + co + 0];
    v4f x3 = r4[kt * 8 + 4 + co + 1];
#pragma unroll
    for (int j = 0; j < 4; ++j) {
      ss += x0[j] * x0[j] + x1[j] * x1[j] + x2[j] * x2[j] + x3[j] * x3[j];
      a[kt][j]      = (_Float16)(x0[j] * XSCALE);
      a[kt][4 + j]  = (_Float16)(x1[j] * XSCALE);
      a[kt][8 + j]  = (_Float16)(x2[j] * XSCALE);
      a[kt][12 + j] = (_Float16)(x3[j] * XSCALE);
    }
  }
  ss += __shfl_xor(ss, 16, 32);  // lane pair (l, l^16) covers the full 128-wide row
  hOut = ss * HSCALE;
}

// Prefetch the A rows of the next grid-stride strip (lowers to global_prefetch_b8).
__device__ __forceinline__ void prefetch_A(const float* __restrict__ X, int row0, int lane) {
  const float* p = X + ((size_t)row0 + (lane & 15)) * DQK + (((lane >> 4) & 1) << 6);
  __builtin_prefetch(p, 0, 0);
  __builtin_prefetch(p + 32, 0, 0);
}

// Batch-load all 4 B tiles for one nt column (lets the compiler issue 8 ds_load_b128
// up front and overlap the waits with the WMMA chain).
__device__ __forceinline__ void load_B(const _Float16* __restrict__ om, int nt, int lane,
                                       v16h (&b)[KT_TILES]) {
#pragma unroll
  for (int kt = 0; kt < KT_TILES; ++kt)
    b[kt] = *(const v16h*)(om + (((kt << 4) + nt) * 32 + lane) * 16);
}

// Split-accumulator 128-deep dot: identical structure in every phase so that
// exp(U - rowmax) is computed against bitwise-identical U values.
__device__ __forceinline__ v8f u_full(const v16h (&a)[KT_TILES], const v16h (&b)[KT_TILES]) {
  v8f u0{}; v8f u1{};
  u0 = __builtin_amdgcn_wmma_f32_16x16x32_f16(false, a[0], false, b[0], (short)0, u0, false, false);
  u1 = __builtin_amdgcn_wmma_f32_16x16x32_f16(false, a[1], false, b[1], (short)0, u1, false, false);
  u0 = __builtin_amdgcn_wmma_f32_16x16x32_f16(false, a[2], false, b[2], (short)0, u0, false, false);
  u1 = __builtin_amdgcn_wmma_f32_16x16x32_f16(false, a[3], false, b[3], (short)0, u1, false, false);
  return u0 + u1;
}

// PHASE 0: global max of U_K.  PHASE 1: w[N], rowmaxQ[N], per-block sumK partials.
// PHASE 2: norm + output.
template <int PHASE>
__global__ __launch_bounds__(TPB) void spooky_phase(
    const float* __restrict__ Q, const float* __restrict__ K,
    const float* __restrict__ V, const float* __restrict__ omega,
    float* __restrict__ out, float* __restrict__ ws, int nRows) {
  extern __shared__ char smem_raw[];
  _Float16* om = (_Float16*)smem_raw;                          // 64 KB, B-layout swizzled
  float* red = (float*)(smem_raw + DQK * MFEAT * sizeof(unsigned short));
  float* skw = red + WPB;                                      // [WPB][256] partials / sumK cache

  const int tid   = threadIdx.x;
  const int lane  = tid & 31;
  const int wv    = tid >> 5;
  const int half8 = (lane >> 4) << 3;  // 0 or 8: C-layout row offset of this half-wave
  const int gstep = gridDim.x * WPB;

  // Stage omega into LDS, pre-swizzled to the CDNA5 16-bit B layout:
  // for tile (kt,nt), lane l needs B[k = kt*32 + (j>>3)*16 + (l>>4)*8 + (j&7)][n = nt*16 + (l&15)], j=0..15
  for (int idx = tid; idx < DQK * MFEAT; idx += TPB) {
    const int k = idx >> 8;
    const int n = idx & 255;
    const int dst = ((((k >> 5) << 4) + (n >> 4)) * 32 + (((k >> 3) & 1) << 4) + (n & 15)) * 16 + (((k >> 4) & 1) << 3) + (k & 7);
    om[dst] = (_Float16)omega[idx];
  }
  if (PHASE == 1) {
    for (int i = tid; i < WPB * MFEAT; i += TPB) skw[i] = 0.0f;
  }
  if (PHASE == 2) {
    if (tid < MFEAT) skw[tid] = ws[WS_SUMK + tid];
  }
  __syncthreads();

  const int nStrips = nRows >> 4;

  if (PHASE == 0) {
    float lm = NEG_BIG;
    for (int strip = blockIdx.x * WPB + wv; strip < nStrips; strip += gstep) {
      const int row0 = strip << 4;
      if (strip + gstep < nStrips) prefetch_A(K, (strip + gstep) << 4, lane);
      v16h aK[KT_TILES]; float hK;
      load_A(K, row0, lane, aK, hK); (void)hK;
      for (int nt = 0; nt < NT_TILES; ++nt) {
        v16h b[KT_TILES];
        load_B(om, nt, lane, b);
        v8f u = u_full(aK, b);
#pragma unroll
        for (int r = 0; r < 8; ++r) lm = fmaxf(lm, u[r]);
      }
    }
    lm = fmaxf(lm, __shfl_xor(lm, 16, 32));
    lm = redmax16(lm);
    if (lane == 0) red[wv] = lm;
    __syncthreads();
    if (tid == 0) {
      float bm = red[0];
#pragma unroll
      for (int i = 1; i < WPB; ++i) bm = fmaxf(bm, red[i]);
      atomic_max_f32(&ws[WS_MAXK], bm);  // max is order-independent -> deterministic
    }
    return;
  }

  if (PHASE == 1) {
    const float maxK = ws[WS_MAXK];
    float* wbuf  = ws + WS_W;
    float* rmbuf = ws + WS_W + nRows;
    float csum[NT_TILES];
#pragma unroll
    for (int nt = 0; nt < NT_TILES; ++nt) csum[nt] = 0.0f;

    for (int strip = blockIdx.x * WPB + wv; strip < nStrips; strip += gstep) {
      const int row0 = strip << 4;
      if (strip + gstep < nStrips) {
        prefetch_A(Q, (strip + gstep) << 4, lane);
        prefetch_A(K, (strip + gstep) << 4, lane);
      }
      v16h aQ[KT_TILES], aK[KT_TILES];
      float hQf, hKf;
      load_A(Q, row0, lane, aQ, hQf);
      load_A(K, row0, lane, aK, hKf);
      float hQc[8], hKc[8];
#pragma unroll
      for (int r = 0; r < 8; ++r) {
        hQc[r] = __shfl(hQf, half8 + r, 32);   // broadcast h into the C layout
        hKc[r] = __shfl(hKf, half8 + r, 32);
      }
      // pass A: row max of U_Q
      float rm[8];
#pragma unroll
      for (int r = 0; r < 8; ++r) rm[r] = NEG_BIG;
      for (int nt = 0; nt < NT_TILES; ++nt) {
        v16h b[KT_TILES];
        load_B(om, nt, lane, b);
        v8f u = u_full(aQ, b);
#pragma unroll
        for (int r = 0; r < 8; ++r) rm[r] = fmaxf(rm[r], u[r]);
      }
#pragma unroll
      for (int r = 0; r < 8; ++r) rm[r] = redmax16(rm[r]);

      // pass B: recompute U_Q, U_K sharing each B tile; w = Qp.Kp ; column sums of Kp
      float wp[8];
#pragma unroll
      for (int r = 0; r < 8; ++r) wp[r] = 0.0f;
      for (int nt = 0; nt < NT_TILES; ++nt) {
        v16h b[KT_TILES];
        load_B(om, nt, lane, b);
        // four independent WMMA chains (uq0,uq1,uk0,uk1) sharing the B tiles
        v8f uq0{}; v8f uq1{}; v8f uk0{}; v8f uk1{};
        uq0 = __builtin_amdgcn_wmma_f32_16x16x32_f16(false, aQ[0], false, b[0], (short)0, uq0, false, false);
        uk0 = __builtin_amdgcn_wmma_f32_16x16x32_f16(false, aK[0], false, b[0], (short)0, uk0, false, false);
        uq1 = __builtin_amdgcn_wmma_f32_16x16x32_f16(false, aQ[1], false, b[1], (short)0, uq1, false, false);
        uk1 = __builtin_amdgcn_wmma_f32_16x16x32_f16(false, aK[1], false, b[1], (short)0, uk1, false, false);
        uq0 = __builtin_amdgcn_wmma_f32_16x16x32_f16(false, aQ[2], false, b[2], (short)0, uq0, false, false);
        uk0 = __builtin_amdgcn_wmma_f32_16x16x32_f16(false, aK[2], false, b[2], (short)0, uk0, false, false);
        uq1 = __builtin_amdgcn_wmma_f32_16x16x32_f16(false, aQ[3], false, b[3], (short)0, uq1, false, false);
        uk1 = __builtin_amdgcn_wmma_f32_16x16x32_f16(false, aK[3], false, b[3], (short)0, uk1, false, false);
        v8f uq = uq0 + uq1;
        v8f uk = uk0 + uk1;
        float cs = 0.0f;
#pragma unroll
        for (int r = 0; r < 8; ++r) {
          float qp = (__expf(uq[r] - hQc[r] - rm[r]) + 1e-4f) * INV_SQRT_M;
          float kp = (__expf(uk[r] - hKc[r] - maxK) + 1e-4f) * INV_SQRT_M;
          wp[r] += qp * kp;
          cs += kp;
        }
        cs += __shfl_xor(cs, 16, 32);  // fold both half-waves per column
        csum[nt] += cs;                // register accumulation across strips (deterministic)
      }
#pragma unroll
      for (int r = 0; r < 8; ++r) wp[r] = redsum16(wp[r]);
#pragma unroll
      for (int r = 0; r < 8; ++r) {
        if (lane == 0)  { wbuf[row0 + r]     = wp[r]; rmbuf[row0 + r]     = rm[r]; }
        if (lane == 16) { wbuf[row0 + 8 + r] = wp[r]; rmbuf[row0 + 8 + r] = rm[r]; }
      }
    }
    // flush this wave's column sums, then one deterministic per-block partial
#pragma unroll
    for (int nt = 0; nt < NT_TILES; ++nt)
      if (lane < 16) skw[wv * MFEAT + nt * 16 + lane] = csum[nt];
    __syncthreads();
    float s = 0.0f;
#pragma unroll
    for (int w = 0; w < WPB; ++w) s += skw[w * MFEAT + tid];
    (ws + WS_W + 2 * (size_t)nRows)[(size_t)blockIdx.x * MFEAT + tid] = s;
    return;
  }

  if (PHASE == 2) {
    const float* wbuf  = ws + WS_W;
    const float* rmbuf = ws + WS_W + nRows;
    for (int strip = blockIdx.x * WPB + wv; strip < nStrips; strip += gstep) {
      const int row0 = strip << 4;
      if (strip + gstep < nStrips) prefetch_A(Q, (strip + gstep) << 4, lane);
      v16h aQ[KT_TILES]; float hQf;
      load_A(Q, row0, lane, aQ, hQf);
      float hQc[8], rm[8];
#pragma unroll
      for (int r = 0; r < 8; ++r) {
        hQc[r] = __shfl(hQf, half8 + r, 32);
        rm[r]  = rmbuf[row0 + half8 + r];      // row max computed in phase 1
      }
      float np[8];
#pragma unroll
      for (int r = 0; r < 8; ++r) np[r] = 0.0f;
      for (int nt = 0; nt < NT_TILES; ++nt) {
        v16h b[KT_TILES];
        load_B(om, nt, lane, b);
        v8f uq = u_full(aQ, b);
        const float skv = skw[nt * 16 + (lane & 15)];
#pragma unroll
        for (int r = 0; r < 8; ++r)
          np[r] += (__expf(uq[r] - hQc[r] - rm[r]) + 1e-4f) * INV_SQRT_M * skv;
      }
#pragma unroll
      for (int r = 0; r < 8; ++r) np[r] = redsum16(np[r]);

      float coeff[8];
#pragma unroll
      for (int r = 0; r < 8; ++r)
        coeff[r] = wbuf[row0 + half8 + r] / (np[r] + 1e-8f);

      const v4f* V4 = (const v4f*)V;
      v4f* O4 = (v4f*)out;
#pragma unroll
      for (int mm = 0; mm < 16; ++mm) {
        float c = __shfl(coeff[mm & 7], (mm < 8) ? 0 : 16, 32);
        size_t off = (size_t)(row0 + mm) * (DQK / 4) + lane;
        O4[off] = V4[off] * c;
      }
    }
    return;
  }
}

__global__ void init_ws(float* ws) {
  if (threadIdx.x == 0) ws[WS_MAXK] = NEG_BIG;
  if (threadIdx.x < MFEAT) ws[WS_SUMK + threadIdx.x] = 0.0f;
}

__global__ void reduce_sumk(float* ws, int nBlocks, int nRows) {
  const int t = threadIdx.x;
  const float* partials = ws + WS_W + 2 * (size_t)nRows;
  float s = 0.0f;
  for (int b = 0; b < nBlocks; ++b) s += partials[(size_t)b * MFEAT + t];  // fixed order
  ws[WS_SUMK + t] = s;
}

extern "C" void kernel_launch(void* const* d_in, const int* in_sizes, int n_in,
                              void* d_out, int out_size, void* d_ws, size_t ws_size,
                              hipStream_t stream) {
  if (n_in < 4) return;
  const float* Q     = (const float*)d_in[0];
  const float* K     = (const float*)d_in[1];
  const float* V     = (const float*)d_in[2];
  const float* omega = (const float*)d_in[3];
  float* out = (float*)d_out;
  float* ws  = (float*)d_ws;
  (void)out_size; (void)ws_size;

  const int nRows   = in_sizes[0] / DQK;     // 262144
  const int nStrips = nRows >> 4;            // 16384
  int nBlocks = (nStrips + WPB * SPW - 1) / (WPB * SPW);  // 512: ~4 strips/wave
  if (nBlocks < 1) nBlocks = 1;
  const size_t shbytes = (size_t)DQK * MFEAT * sizeof(unsigned short)  // omega f16
                       + (size_t)(WPB + WPB * MFEAT) * sizeof(float);  // red + partials

  init_ws<<<1, TPB, 0, stream>>>(ws);
  spooky_phase<0><<<nBlocks, TPB, shbytes, stream>>>(Q, K, V, omega, out, ws, nRows);
  spooky_phase<1><<<nBlocks, TPB, shbytes, stream>>>(Q, K, V, omega, out, ws, nRows);
  reduce_sumk<<<1, TPB, 0, stream>>>(ws, nBlocks, nRows);
  spooky_phase<2><<<nBlocks, TPB, shbytes, stream>>>(Q, K, V, omega, out, ws, nRows);
}
